// GraphAttentionalLayer_84447646974686
// MI455X (gfx1250) — matrix operand order repaired and checked by reading, not compiled, not run on hardware
//
#include <hip/hip_runtime.h>
#include <hip/hip_bf16.h>
#include <math.h>

// ---------------------------------------------------------------------------
// GAT layer for MI455X (gfx1250, wave32, WMMA bf16 16x16x32)
// Pipeline: cvt_h, cvt_w -> gat_gemm (ht = h@W, stored transposed bf16)
//           -> gat_srcdst (src/dst = ht . a)
//           -> gat_stats  (C[h][i] = rowmax + log(rowsumexp), streams adj once)
//           -> gat_attn   (flash-style P@V with WMMA, mean over heads)
// ---------------------------------------------------------------------------

typedef __attribute__((ext_vector_type(16))) __bf16 v16bf;
typedef __attribute__((ext_vector_type(8)))  float  v8f;
typedef unsigned short u16;
typedef unsigned int   u32;

#define N_NODES 8192
#define F_IN    512
#define NHEADS  4
#define NHID    512
#define NCOLS   (NHEADS * NHID)   // 2048

union V16 { v16bf v; u16 s[16]; uint4 q[2]; };

__device__ __forceinline__ u16 f2bf(float f) {
  u32 u = __builtin_bit_cast(u32, f);
  u32 r = u + 0x7fffu + ((u >> 16) & 1u);   // round-to-nearest-even
  return (u16)(r >> 16);
}
__device__ __forceinline__ float bf2f(u16 b) {
  u32 u = ((u32)b) << 16;
  return __builtin_bit_cast(float, u);
}

// ---- fp32 -> bf16 conversions --------------------------------------------
__global__ __launch_bounds__(256) void cvt_h(const float* __restrict__ h,
                                             u16* __restrict__ hb) {
  int g = blockIdx.x * 256 + threadIdx.x;
  hb[g] = f2bf(h[g]);
}
// Wt[n][k] = bf16(W[k][n])  (transpose so GEMM B-fragments are contiguous)
__global__ __launch_bounds__(256) void cvt_w(const float* __restrict__ W,
                                             u16* __restrict__ Wt) {
  int g = blockIdx.x * 256 + threadIdx.x;   // g = n*512 + k
  int n = g >> 9, k = g & 511;
  Wt[g] = f2bf(W[k * NCOLS + n]);
}

// ---- ht = h @ W, stored transposed: htT[(head*512+d)*8192 + node] ---------
__global__ __launch_bounds__(256) void gat_gemm(const u16* __restrict__ hb,
                                                const u16* __restrict__ Wt,
                                                u16* __restrict__ htT) {
  const int t = threadIdx.x, lane = t & 31, w = t >> 5;
  const int m0 = blockIdx.x * 16;             // node tile
  const int n0 = blockIdx.y * 128 + w * 16;   // output-column tile (per wave)
  const int lm = lane & 15, hi = lane >> 4;
  const int c0 = hi ? 8 : 0;                  // A-fragment K chunk base
  const int kb = hi ? 16 : 0;                 // unused; B fragment also uses c0
  const u16* arow = hb + (m0 + lm) * F_IN;
  const u16* brow = Wt + (n0 + lm) * F_IN;

  v8f acc = {};
  for (int k0 = 0; k0 < F_IN; k0 += 32) {
    V16 a, b;
    a.q[0] = *(const uint4*)(arow + k0 + c0);        // K = c0..c0+7
    a.q[1] = *(const uint4*)(arow + k0 + c0 + 16);   // K = c0+16..c0+23
    b.q[0] = *(const uint4*)(brow + k0 + c0);        // K = c0..c0+7
    b.q[1] = *(const uint4*)(brow + k0 + c0 + 16);   // K = c0+16..c0+23
    acc = __builtin_amdgcn_wmma_f32_16x16x32_bf16(
        false, a.v, false, b.v, (short)0, acc, false, false);
  }
  // C layout: lane owns column (n0+lm), rows m0 + 8*hi + r  -> transposed
  // store is 8 consecutive bf16 (16B) per lane.
  int col = n0 + lm;
  int head = col >> 9, dd = col & (NHID - 1);
  alignas(16) u16 ov[8];
#pragma unroll
  for (int r = 0; r < 8; ++r) ov[r] = f2bf(acc[r]);
  u16* dst = htT + (head * NHID + dd) * N_NODES + m0 + 8 * hi;
  *(uint4*)dst = *(const uint4*)ov;
}

// ---- src[h][i] = ht[h][i][:] . a_src[h],  dst likewise --------------------
__global__ __launch_bounds__(256) void gat_srcdst(const u16* __restrict__ htT,
                                                  const float* __restrict__ a,
                                                  float* __restrict__ srcp,
                                                  float* __restrict__ dstp) {
  int gid = blockIdx.x * 256 + threadIdx.x;   // h*8192 + i
  int h = gid >> 13, i = gid & (N_NODES - 1);
  float sa = 0.f, sd = 0.f;
  for (int d = 0; d < NHID; ++d) {
    float v = bf2f(htT[(h * NHID + d) * N_NODES + i]);  // coalesced over i
    sa += v * a[h * 1024 + d];
    sd += v * a[h * 1024 + 512 + d];
  }
  srcp[gid] = sa;
  dstp[gid] = sd;
}

// ---- C[h][i] = rowmax + log(rowsumexp); one wave per row, 4 heads fused ---
__global__ __launch_bounds__(256) void gat_stats(const float* __restrict__ adj,
                                                 const float* __restrict__ srcp,
                                                 const float* __restrict__ dstp,
                                                 float* __restrict__ Cp) {
  const int lane = threadIdx.x & 31;
  const int i = blockIdx.x * 8 + (threadIdx.x >> 5);
  float si[NHEADS];
#pragma unroll
  for (int h = 0; h < NHEADS; ++h) si[h] = srcp[h * N_NODES + i];

  float m[NHEADS] = {-3.0e38f, -3.0e38f, -3.0e38f, -3.0e38f};
  for (int j = lane; j < N_NODES; j += 32) {
    float av = adj[i * N_NODES + j];
#pragma unroll
    for (int h = 0; h < NHEADS; ++h) {
      float x = si[h] + dstp[h * N_NODES + j];
      float lr = x > 0.f ? x : 0.01f * x;
      m[h] = fmaxf(m[h], lr + av);
    }
  }
#pragma unroll
  for (int h = 0; h < NHEADS; ++h)
    for (int off = 16; off > 0; off >>= 1)
      m[h] = fmaxf(m[h], __shfl_xor(m[h], off, 32));

  float l[NHEADS] = {0.f, 0.f, 0.f, 0.f};
  for (int j = lane; j < N_NODES; j += 32) {   // 2nd pass hits WGP$/L2
    float av = adj[i * N_NODES + j];
#pragma unroll
    for (int h = 0; h < NHEADS; ++h) {
      float x = si[h] + dstp[h * N_NODES + j];
      float lr = x > 0.f ? x : 0.01f * x;
      l[h] += __expf(lr + av - m[h]);
    }
  }
#pragma unroll
  for (int h = 0; h < NHEADS; ++h)
    for (int off = 16; off > 0; off >>= 1) l[h] += __shfl_xor(l[h], off, 32);

  if (lane == 0) {
#pragma unroll
    for (int h = 0; h < NHEADS; ++h) Cp[h * N_NODES + i] = m[h] + __logf(l[h]);
  }
}

// ---- fused attention: out[i][d] = 0.25 * sum_h sum_j P_h[i][j] * ht[h][j][d]
// Block: 64 rows x all 512 d; 8 waves, wave w owns d in [w*64, w*64+64).
// Per j-tile (32 cols): adj cached in regs, reused across 4 heads; P built
// cooperatively in LDS (one exp per element), consumed as WMMA A-fragments.
#define PLPITCH 40   // u16 per LDS row (80B) -> conflict-free b128 reads
__global__ __launch_bounds__(256) void gat_attn(const float* __restrict__ adj,
                                                const u16* __restrict__ htT,
                                                const float* __restrict__ srcp,
                                                const float* __restrict__ dstp,
                                                const float* __restrict__ Cp,
                                                float* __restrict__ out) {
  __shared__ __align__(16) u16 PL[64 * PLPITCH];
  __shared__ float2 rowc[NHEADS * 64];   // {src_i, C_i}

  const int t = threadIdx.x, lane = t & 31, w = t >> 5;
  const int i0 = blockIdx.x * 64;
  const int lm = lane & 15, hi = lane >> 4;
  const int c0 = hi ? 8 : 0;    // A-fragment K chunk base
  const int kb = hi ? 16 : 0;   // unused; B fragment also uses c0
  const int cc = lane;          // P-compute: column in j-tile
  const int r0 = w;             // P-compute: base row (rows r0 + 8k)

  {  // stage per-row constants: 256 threads <-> 4 heads * 64 rows
    int h = t >> 6, r = t & 63;
    rowc[t] = make_float2(srcp[h * N_NODES + i0 + r], Cp[h * N_NODES + i0 + r]);
  }
  __syncthreads();

  v8f acc[4][4];
#pragma unroll
  for (int rt = 0; rt < 4; ++rt)
#pragma unroll
    for (int dt = 0; dt < 4; ++dt) acc[rt][dt] = (v8f){};

  for (int j0 = 0; j0 < N_NODES; j0 += 32) {
    float adjv[8];
#pragma unroll
    for (int k = 0; k < 8; ++k)   // adj tile once, reused by all 4 heads
      adjv[k] = adj[(i0 + r0 + 8 * k) * N_NODES + j0 + cc];

#pragma unroll 1
    for (int h = 0; h < NHEADS; ++h) {
      float dstv = dstp[h * N_NODES + j0 + cc];
      __syncthreads();   // previous P consumers done
#pragma unroll
      for (int k = 0; k < 8; ++k) {
        float2 rc = rowc[h * 64 + r0 + 8 * k];
        float x = rc.x + dstv;
        float lr = x > 0.f ? x : 0.01f * x;
        // P = exp(leaky + adj - (m + log l)) = softmax weight, in [0,1]
        PL[(r0 + 8 * k) * PLPITCH + cc] = f2bf(__expf(lr + adjv[k] - rc.y));
      }
      __syncthreads();

      V16 afr[4];
#pragma unroll
      for (int rt = 0; rt < 4; ++rt) {
        const u16* ap = &PL[(rt * 16 + lm) * PLPITCH + c0];
        afr[rt].q[0] = *(const uint4*)ap;          // K = c0..c0+7
        afr[rt].q[1] = *(const uint4*)(ap + 16);   // K = c0+16..c0+23
      }
#pragma unroll
      for (int dt = 0; dt < 4; ++dt) {
        V16 b;
        const u16* bp =
            &htT[(h * NHID + w * 64 + dt * 16 + lm) * N_NODES + j0 + c0];
        b.q[0] = *(const uint4*)bp;
        b.q[1] = *(const uint4*)(bp + 16);
#pragma unroll
        for (int rt = 0; rt < 4; ++rt)
          acc[rt][dt] = __builtin_amdgcn_wmma_f32_16x16x32_bf16(
              false, afr[rt].v, false, b.v, (short)0, acc[rt][dt], false, false);
      }
    }
  }

  // mean over heads (1/l already folded into P)
#pragma unroll
  for (int rt = 0; rt < 4; ++rt)
#pragma unroll
    for (int dt = 0; dt < 4; ++dt) {
      int dcol = w * 64 + dt * 16 + lm;
#pragma unroll
      for (int r = 0; r < 8; ++r) {
        int i = i0 + rt * 16 + 8 * hi + r;
        out[i * NHID + dcol] = acc[rt][dt][r] * 0.25f;
      }
    }
}

// ---------------------------------------------------------------------------
extern "C" void kernel_launch(void* const* d_in, const int* in_sizes, int n_in,
                              void* d_out, int out_size, void* d_ws,
                              size_t ws_size, hipStream_t stream) {
  const float* h   = (const float*)d_in[0];   // (8192, 512)
  const float* adj = (const float*)d_in[1];   // (8192, 8192)
  const float* W   = (const float*)d_in[2];   // (512, 2048)
  const float* a   = (const float*)d_in[3];   // (4, 1024, 1)
  float* out = (float*)d_out;                 // (8192, 512) fp32

  char* ws = (char*)d_ws;
  u16*   hb   = (u16*)(ws);                                  //  8 MB
  u16*   Wt   = (u16*)(ws + (8ull << 20));                   //  2 MB
  u16*   htT  = (u16*)(ws + (10ull << 20));                  // 32 MB
  float* srcp = (float*)(ws + (42ull << 20));                // 128 KB
  float* dstp = (float*)(ws + (42ull << 20) + (128u << 10)); // 128 KB
  float* Cp   = (float*)(ws + (42ull << 20) + (256u << 10)); // 128 KB

  cvt_h<<<N_NODES * F_IN / 256, 256, 0, stream>>>(h, hb);
  cvt_w<<<NCOLS * F_IN / 256, 256, 0, stream>>>(W, Wt);
  gat_gemm<<<dim3(N_NODES / 16, NCOLS / 128), 256, 0, stream>>>(hb, Wt, htT);
  gat_srcdst<<<NHEADS * N_NODES / 256, 256, 0, stream>>>(htT, a, srcp, dstp);
  gat_stats<<<N_NODES / 8, 256, 0, stream>>>(adj, srcp, dstp, Cp);
  gat_attn<<<N_NODES / 64, 256, 0, stream>>>(adj, htT, srcp, dstp, Cp, out);
}
